// FALayer_52226802319650
// MI455X (gfx1250) — matrix operand order repaired and checked by reading, not compiled, not run on hardware
//
#include <hip/hip_runtime.h>
#include <math.h>

typedef __attribute__((ext_vector_type(16))) _Float16 v16h;
typedef __attribute__((ext_vector_type(8)))  float    v8f;

#define DIM   64
#define TWO_D 128

// ---------------------------------------------------------------------------
// Zero-init z accumulator (float4 stores)
// ---------------------------------------------------------------------------
__global__ void zero_f4_kernel(float4* __restrict__ p, long n4) {
  long i = (long)blockIdx.x * blockDim.x + threadIdx.x;
  if (i < n4) p[i] = make_float4(0.f, 0.f, 0.f, 0.f);
}

// ---------------------------------------------------------------------------
// Per-node projections: lowd/lows/highd/highs = h . {Wl[:64],Wl[64:],Wh[:64],Wh[64:]}
// One wave (32 lanes) per node, each lane covers 2 dims, xor-shuffle reduce.
// ---------------------------------------------------------------------------
__global__ void __launch_bounds__(256)
proj_kernel(const float* __restrict__ h,
            const float* __restrict__ Wl, const float* __restrict__ Wh,
            float* __restrict__ lowd,  float* __restrict__ lows,
            float* __restrict__ highd, float* __restrict__ highs, int N_) {
  int wave = (int)((blockIdx.x * blockDim.x + threadIdx.x) >> 5);
  int lane = threadIdx.x & 31;
  if (wave >= N_) return;

  float2 hv  = ((const float2*)(h + (long)wave * DIM))[lane];
  float2 wld = ((const float2*)Wl)[lane];
  float2 wls = ((const float2*)Wl)[32 + lane];
  float2 whd = ((const float2*)Wh)[lane];
  float2 whs = ((const float2*)Wh)[32 + lane];

  float pld = hv.x * wld.x + hv.y * wld.y;
  float pls = hv.x * wls.x + hv.y * wls.y;
  float phd = hv.x * whd.x + hv.y * whd.y;
  float phs = hv.x * whs.x + hv.y * whs.y;

  #pragma unroll
  for (int off = 16; off > 0; off >>= 1) {
    pld += __shfl_xor(pld, off, 32);
    pls += __shfl_xor(pls, off, 32);
    phd += __shfl_xor(phd, off, 32);
    phs += __shfl_xor(phs, off, 32);
  }
  if (lane == 0) {
    lowd[wave] = pld;  lows[wave]  = pls;
    highd[wave] = phd; highs[wave] = phs;
  }
}

// ---------------------------------------------------------------------------
// Edge stage: wave per edge. Gates computed (uniformly) per lane, then each
// lane scatter-adds 2 dims of z_low and 2 dims of z_high with native f32
// global atomics. z layout: [N][128] = [z_low(64) | z_high(64)].
// ---------------------------------------------------------------------------
__global__ void __launch_bounds__(256)
edge_kernel(const float* __restrict__ h, const float* __restrict__ dnorm,
            const int* __restrict__ src, const int* __restrict__ dst,
            const float* __restrict__ lowd,  const float* __restrict__ lows,
            const float* __restrict__ highd, const float* __restrict__ highs,
            const float* __restrict__ bl, const float* __restrict__ bh,
            float* __restrict__ z, int E_) {
  int e = (int)((blockIdx.x * blockDim.x + threadIdx.x) >> 5);
  if (e >= E_) return;
  int lane = threadIdx.x & 31;

  int s = src[e];
  int t = dst[e];

  float xl = lowd[t] + lows[s] + bl[0];
  float xh = highd[t] + highs[s] + bh[0];
  // where(x>0, x, -0.5x) is already >= 0, so relu() is identity
  xl = (xl > 0.f) ? xl : (-0.5f * xl);
  xh = (xh > 0.f) ? xh : (-0.5f * xh);
  float dd = dnorm[t] * dnorm[s];
  float el = tanhf(xl) * dd;
  float eh = tanhf(-xh) * dd;

  float2 hv = ((const float2*)(h + (long)s * DIM))[lane];
  float* zr = z + (long)t * TWO_D;
  unsafeAtomicAdd(zr + 2 * lane,           hv.x * el);
  unsafeAtomicAdd(zr + 2 * lane + 1,       hv.y * el);
  unsafeAtomicAdd(zr + DIM + 2 * lane,     hv.x * eh);
  unsafeAtomicAdd(zr + DIM + 2 * lane + 1, hv.y * eh);
}

// ---------------------------------------------------------------------------
// Pack Wr.T [128,64] (f32, Wrt[k][j] = Wr[j*128+k]) into f16 WMMA B-operand
// layout: bp[((nTile*4 + kTile)*32 + lane)*16 + elem],
// elem e of lane L = Wrt[kTile*32 + (e>>3)*16 + (L>>4)*8 + (e&7)][nTile*16 + (L&15)]
// ---------------------------------------------------------------------------
__global__ void bpack_kernel(const float* __restrict__ Wr, _Float16* __restrict__ bp) {
  int i = blockIdx.x * blockDim.x + threadIdx.x;   // 4*4*32*16 = 8192
  if (i >= 8192) return;
  int e    = i & 15;
  int lane = (i >> 4) & 31;
  int kt   = (i >> 9) & 3;
  int nt   = (i >> 11) & 3;
  int k    = kt * 32 + (e >> 3) * 16 + (lane >> 4) * 8 + (e & 7);
  int col  = nt * 16 + (lane & 15);
  bp[i] = (_Float16)Wr[col * TWO_D + k];           // Wrt[k][col]
}

// ---------------------------------------------------------------------------
// out = z[N,128] @ Wr.T[128,64] + br via v_wmma_f32_16x16x32_f16.
// One wave per 16x16 output tile; K=128 -> 4 WMMA. A converted f32->f16
// on the fly following the 16-bit A 16x32 lane layout.
// ---------------------------------------------------------------------------
__global__ void __launch_bounds__(256)
gemm_kernel(const float* __restrict__ z, const _Float16* __restrict__ bp,
            const float* __restrict__ br, float* __restrict__ out, int N_) {
  int w    = (int)((blockIdx.x * blockDim.x + threadIdx.x) >> 5);
  int lane = threadIdx.x & 31;
  int mTile = w >> 2;
  int nTile = w & 3;
  if (mTile * 16 >= N_) return;          // wave-uniform: EXEC stays all-1s

  int half = lane >> 4;
  int m    = mTile * 16 + (lane & 15);
  const float* zrow = z + (long)m * TWO_D + half * 8;

  v8f c = {};
  #pragma unroll
  for (int kt = 0; kt < 4; ++kt) {
    const float4* p = (const float4*)(zrow + kt * 32);
    float4 a0 = p[0], a1 = p[1];   // K = kt*32 + half*8 + 0..7
    float4 a2 = p[4], a3 = p[5];   // K = kt*32 + 16 + half*8 + 0..7
    v16h a;
    a[0]  = (_Float16)a0.x; a[1]  = (_Float16)a0.y;
    a[2]  = (_Float16)a0.z; a[3]  = (_Float16)a0.w;
    a[4]  = (_Float16)a1.x; a[5]  = (_Float16)a1.y;
    a[6]  = (_Float16)a1.z; a[7]  = (_Float16)a1.w;
    a[8]  = (_Float16)a2.x; a[9]  = (_Float16)a2.y;
    a[10] = (_Float16)a2.z; a[11] = (_Float16)a2.w;
    a[12] = (_Float16)a3.x; a[13] = (_Float16)a3.y;
    a[14] = (_Float16)a3.z; a[15] = (_Float16)a3.w;

    v16h b = *(const v16h*)(bp + (((nTile * 4 + kt) * 32 + lane) << 4));

    c = __builtin_amdgcn_wmma_f32_16x16x32_f16(
        /*neg_a=*/false, a, /*neg_b=*/false, b,
        /*c_mod=*/(short)0, c, /*reuse_a=*/false, /*reuse_b=*/false);
  }

  int col = nTile * 16 + (lane & 15);
  float bias = br[col];
  #pragma unroll
  for (int v = 0; v < 8; ++v) {
    out[(long)(mTile * 16 + v + 8 * half) * DIM + col] = c[v] + bias;
  }
}

// ---------------------------------------------------------------------------
// Host-side launch
// ---------------------------------------------------------------------------
extern "C" void kernel_launch(void* const* d_in, const int* in_sizes, int n_in,
                              void* d_out, int out_size, void* d_ws, size_t ws_size,
                              hipStream_t stream) {
  const float* h   = (const float*)d_in[0];
  const float* dnm = (const float*)d_in[1];
  const int*   src = (const int*)  d_in[2];
  const int*   dst = (const int*)  d_in[3];
  const float* Wl  = (const float*)d_in[4];
  const float* bl  = (const float*)d_in[5];
  const float* Wh  = (const float*)d_in[6];
  const float* bh  = (const float*)d_in[7];
  const float* Wr  = (const float*)d_in[8];
  const float* br  = (const float*)d_in[9];
  float* out = (float*)d_out;

  int N = in_sizes[1];   // d is [N]
  int E = in_sizes[2];   // src is [E]

  char* ws = (char*)d_ws;
  float*    z    = (float*)ws;                               // N*128 f32
  float*    proj = (float*)(ws + (size_t)N * TWO_D * 4);     // 4*N f32
  float*    lowd  = proj;
  float*    lows  = proj + N;
  float*    highd = proj + 2 * (size_t)N;
  float*    highs = proj + 3 * (size_t)N;
  _Float16* bp   = (_Float16*)(ws + (size_t)N * TWO_D * 4 + (size_t)4 * N * 4); // 8192 f16

  // 1) zero z accumulator
  long n4 = (long)N * TWO_D / 4;
  zero_f4_kernel<<<(unsigned)((n4 + 255) / 256), 256, 0, stream>>>((float4*)z, n4);

  // 2) per-node projections (wave per node)
  long projThreads = (long)N * 32;
  proj_kernel<<<(unsigned)((projThreads + 255) / 256), 256, 0, stream>>>(
      h, Wl, Wh, lowd, lows, highd, highs, N);

  // 3) pack Wr.T into WMMA B layout
  bpack_kernel<<<(8192 + 255) / 256, 256, 0, stream>>>(Wr, bp);

  // 4) edge gather/gate/scatter (wave per edge)
  long edgeThreads = (long)E * 32;
  edge_kernel<<<(unsigned)((edgeThreads + 255) / 256), 256, 0, stream>>>(
      h, dnm, src, dst, lowd, lows, highd, highs, bl, bh, z, E);

  // 5) WMMA output GEMM (wave per 16x16 tile)
  long numWaves = (long)((N + 15) / 16) * 4;
  long gemmThreads = numWaves * 32;
  gemm_kernel<<<(unsigned)((gemmThreads + 255) / 256), 256, 0, stream>>>(
      z, bp, br, out, N);
}
